// GIN_70325794504770
// MI455X (gfx1250) — matrix operand order repaired and checked by reading, not compiled, not run on hardware
//
#include <hip/hip_runtime.h>

#define NN 100000
#define EE 1600000
#define FF 64
#define CC 10
#define LL 4
#define BB 128

typedef __attribute__((ext_vector_type(16))) __bf16 v16bf;
typedef __attribute__((ext_vector_type(8)))  __bf16 v8bf;
typedef __attribute__((ext_vector_type(8)))  float  v8f;

__device__ __forceinline__ float sigmoidf(float x) {
    return 1.0f / (1.0f + __expf(-x));
}

// ---------------------------------------------------------------------------
// agg <- h   (vectorized copy; fuses the "+h" self-loop term of GINConv eps=0)
// ---------------------------------------------------------------------------
__global__ void copy_kernel(const float4* __restrict__ src, float4* __restrict__ dst, int n4) {
    int i = blockIdx.x * blockDim.x + threadIdx.x;
    if (i < n4) dst[i] = src[i];
}

__global__ void zero_kernel(float* __restrict__ p, int n) {
    int i = blockIdx.x * blockDim.x + threadIdx.x;
    if (i < n) p[i] = 0.0f;
}

// ---------------------------------------------------------------------------
// Scatter-add: one wave per (edge, 32-feature half); lane = one feature.
// Edge id is wave-uniform -> edge_index loads scalarize; gathers/atomics are
// contiguous 128B per half-wave and L2-resident (h is 25.6MB << 192MB L2).
// ---------------------------------------------------------------------------
__global__ void scatter_kernel(const float* __restrict__ hsrc,
                               const long long* __restrict__ ei,
                               float* __restrict__ agg) {
    int wid  = (blockIdx.x * blockDim.x + threadIdx.x) >> 5;
    int lane = threadIdx.x & 31;
    int e    = wid >> 1;
    if (e >= EE) return;
    int f = ((wid & 1) << 5) + lane;
    int s = (int)ei[e];
    int d = (int)ei[EE + e];
    atomicAdd(agg + (size_t)d * FF + f, hsrc[(size_t)s * FF + f]);
}

// ---------------------------------------------------------------------------
// Pre-swizzle W into A-fragment-ready order for the transposed product
// Y^T = W^T @ X^T.   A[i][k] = W[kt*32 + koff][mt*16 + i]  (W is [F_in,F_out])
// A 16x32 fragment, contiguous k order (same as B): lane l: i = l&15, g = l>>4,
//   a[p]   = A[i][g*16 + p]       p = 0..7
//   a[8+p] = A[i][g*16 + 8 + p]   p = 0..7
// Stored as wswz[((kt*4+mt)*32 + lane)*16 + p] so a fragment is one aligned
// 32-byte per-lane block (2x ds_load_b128 at use).
// One block per weight matrix: blockIdx.x even -> W1[l], odd -> W2[l], l = x>>1.
// ---------------------------------------------------------------------------
__global__ __launch_bounds__(256) void swizzle_weights(const float* __restrict__ W1s,
                                                       const float* __restrict__ W2s,
                                                       __bf16* __restrict__ wswz) {
    int mat = blockIdx.x;
    const float* W = (mat & 1) ? (W2s + (size_t)(mat >> 1) * FF * FF)
                               : (W1s + (size_t)(mat >> 1) * FF * FF);
    __bf16* dst = wswz + (size_t)mat * FF * FF;

    int t    = threadIdx.x;          // 256 threads = 256 fragments-lanes
    int kt   = t >> 7;
    int mt   = (t >> 5) & 3;
    int lane = t & 31;
    int i    = lane & 15;
    int g    = lane >> 4;

    v16bf a;
#pragma unroll
    for (int p = 0; p < 16; p++) {
        int koff = g * 16 + p;
        a[p] = (__bf16)W[(kt * 32 + koff) * FF + mt * 16 + i];
    }
    *(v16bf*)&dst[((size_t)((kt * 4 + mt) * 32 + lane)) * 16] = a;
}

__device__ __forceinline__ v16bf cvt16(float4 f0, float4 f1, float4 f2, float4 f3) {
    v16bf b;
    b[0]  = (__bf16)f0.x; b[1]  = (__bf16)f0.y; b[2]  = (__bf16)f0.z; b[3]  = (__bf16)f0.w;
    b[4]  = (__bf16)f1.x; b[5]  = (__bf16)f1.y; b[6]  = (__bf16)f1.z; b[7]  = (__bf16)f1.w;
    b[8]  = (__bf16)f2.x; b[9]  = (__bf16)f2.y; b[10] = (__bf16)f2.z; b[11] = (__bf16)f2.w;
    b[12] = (__bf16)f3.x; b[13] = (__bf16)f3.y; b[14] = (__bf16)f3.z; b[15] = (__bf16)f3.w;
    return b;
}

// ---------------------------------------------------------------------------
// h = sigmoid(sigmoid(agg @ W1) @ W2), computed as Y^T = W^T @ X^T.
// Block = 256 threads = 8 waves; each wave owns one 16-row tile.
//   A (weights)  : pre-swizzled bf16 fragments, 2x ds_load_b128 each
//   B (acts)     : lane = 16 consecutive features of one row
//                  (global float4 x4 for matmul1, 2x ds_load_b128 for matmul2)
//   D (out^T)    : lane holds 8 consecutive features of its own row ->
//                  vectorized b128 stores; matmul1-D and matmul2-B share lane -> row,
//                  so the inter-matmul reshape is wave-private and contiguous.
// 16x v_wmma_f32_16x16x32_bf16 per tile.
// ---------------------------------------------------------------------------
__global__ __launch_bounds__(256) void mlp_kernel(const float* __restrict__ agg,
                                                  const __bf16* __restrict__ wsw1,
                                                  const __bf16* __restrict__ wsw2,
                                                  float* __restrict__ hout,
                                                  int ntiles) {
    __shared__ __align__(32) __bf16 w1s[FF * FF];
    __shared__ __align__(32) __bf16 w2s[FF * FF];
    __shared__ __align__(32) __bf16 mid[8][16 * FF];

    const int tid = threadIdx.x;
    {   // contiguous vectorized stage of fragment-ready weights into LDS
        const uint4* g1 = (const uint4*)wsw1;
        const uint4* g2 = (const uint4*)wsw2;
        uint4* l1 = (uint4*)w1s;
        uint4* l2 = (uint4*)w2s;
        for (int i = tid; i < FF * FF / 8; i += 256) { l1[i] = g1[i]; l2[i] = g2[i]; }
    }
    __syncthreads();

    const int wave = tid >> 5;
    const int lane = tid & 31;
    int tile = blockIdx.x * 8 + wave;
    if (tile >= ntiles) tile = ntiles - 1;   // tail waves recompute same tile (benign)
    const int row0 = tile * 16;
    const int n    = lane & 15;              // data row within tile
    const int g    = lane >> 4;

    const v16bf* w1f = (const v16bf*)w1s;
    const v16bf* w2f = (const v16bf*)w2s;
    const v8f vzero = {0.f, 0.f, 0.f, 0.f, 0.f, 0.f, 0.f, 0.f};

    v8f acc[4];
#pragma unroll
    for (int mt = 0; mt < 4; mt++) acc[mt] = vzero;

    // ---- matmul 1: mid^T = W1^T @ agg^T ----
    const float* arow = agg + (size_t)(row0 + n) * FF + g * 16;
#pragma unroll
    for (int kt = 0; kt < 2; kt++) {
        const float4* bp = (const float4*)(arow + kt * 32);
        v16bf b = cvt16(bp[0], bp[1], bp[2], bp[3]);
#pragma unroll
        for (int mt = 0; mt < 4; mt++) {
            v16bf a = w1f[(kt * 4 + mt) * 32 + lane];
            acc[mt] = __builtin_amdgcn_wmma_f32_16x16x32_bf16(
                false, a, false, b, (short)0, acc[mt], false, false);
        }
    }

    // sigmoid -> mid row-major bf16; lane stores 8 consecutive feats of its row
#pragma unroll
    for (int mt = 0; mt < 4; mt++) {
        v8bf s;
#pragma unroll
        for (int r = 0; r < 8; r++) s[r] = (__bf16)sigmoidf(acc[mt][r]);
        *(v8bf*)&mid[wave][n * FF + mt * 16 + 8 * g] = s;
    }
    // wave-private exchange (lane <-> lane+16 only): DS-counter wait suffices
    asm volatile("s_wait_dscnt 0x0" ::: "memory");

    // ---- matmul 2: out^T = W2^T @ mid^T ----
#pragma unroll
    for (int mt = 0; mt < 4; mt++) acc[mt] = vzero;
#pragma unroll
    for (int kt = 0; kt < 2; kt++) {
        v16bf b = *(const v16bf*)&mid[wave][n * FF + kt * 32 + g * 16];
#pragma unroll
        for (int mt = 0; mt < 4; mt++) {
            v16bf a = w2f[(kt * 4 + mt) * 32 + lane];
            acc[mt] = __builtin_amdgcn_wmma_f32_16x16x32_bf16(
                false, a, false, b, (short)0, acc[mt], false, false);
        }
    }

    // sigmoid -> global, lane writes 8 consecutive f32 of its own row
    float* orow = hout + (size_t)(row0 + n) * FF;
#pragma unroll
    for (int mt = 0; mt < 4; mt++) {
        float4 o0 = {sigmoidf(acc[mt][0]), sigmoidf(acc[mt][1]),
                     sigmoidf(acc[mt][2]), sigmoidf(acc[mt][3])};
        float4 o1 = {sigmoidf(acc[mt][4]), sigmoidf(acc[mt][5]),
                     sigmoidf(acc[mt][6]), sigmoidf(acc[mt][7])};
        *(float4*)&orow[mt * 16 + 8 * g]     = o0;
        *(float4*)&orow[mt * 16 + 8 * g + 4] = o1;
    }
}

// ---------------------------------------------------------------------------
// global_add_pool: one wave per (node, half-row), lane per feature.
// ---------------------------------------------------------------------------
__global__ void pool_kernel(const float* __restrict__ h,
                            const long long* __restrict__ batch,
                            float* __restrict__ xr) {
    int wid  = (blockIdx.x * blockDim.x + threadIdx.x) >> 5;
    int lane = threadIdx.x & 31;
    int nd   = wid >> 1;
    if (nd >= NN) return;
    int f = ((wid & 1) << 5) + lane;
    int b = (int)batch[nd];
    atomicAdd(xr + (size_t)b * FF + f, h[(size_t)nd * FF + f]);
}

// ---------------------------------------------------------------------------
// logits = xr @ fc_w.T + fc_b; log_softmax. One thread per graph (B=128).
// ---------------------------------------------------------------------------
__global__ void head_kernel(const float* __restrict__ xr,
                            const float* __restrict__ fcw,
                            const float* __restrict__ fcb,
                            float* __restrict__ out) {
    int b = threadIdx.x;
    if (b >= BB) return;
    float logit[CC];
    float mx = -1e30f;
#pragma unroll
    for (int c = 0; c < CC; c++) {
        float s = fcb[c];
        for (int f = 0; f < FF; f++) s += xr[b * FF + f] * fcw[c * FF + f];
        logit[c] = s;
        mx = fmaxf(mx, s);
    }
    float se = 0.0f;
#pragma unroll
    for (int c = 0; c < CC; c++) se += __expf(logit[c] - mx);
    float lse = mx + __logf(se);
#pragma unroll
    for (int c = 0; c < CC; c++) out[b * CC + c] = logit[c] - lse;
}

extern "C" void kernel_launch(void* const* d_in, const int* in_sizes, int n_in,
                              void* d_out, int out_size, void* d_ws, size_t ws_size,
                              hipStream_t stream) {
    const float*     x     = (const float*)d_in[0];
    const long long* ei    = (const long long*)d_in[1];
    const long long* batch = (const long long*)d_in[2];
    const float*     W1s   = (const float*)d_in[3];
    const float*     W2s   = (const float*)d_in[4];
    const float*     fcw   = (const float*)d_in[5];
    const float*     fcb   = (const float*)d_in[6];

    float*  out  = (float*)d_out;
    float*  h    = (float*)d_ws;                         // [N, F]
    float*  agg  = h + (size_t)NN * FF;                  // [N, F]
    __bf16* wswz = (__bf16*)(agg + (size_t)NN * FF);     // [2L, F*F] fragment-ready
    float*  xr   = out + BB * CC;                        // xr slice of output tuple

    const int n4           = NN * FF / 4;
    const int copy_blocks  = (n4 + 255) / 256;
    const int scatter_blks = (EE * 2 + 7) / 8;           // 2 waves/edge, 8 waves/block
    const int ntiles       = (NN + 15) / 16;
    const int mlp_blocks   = (ntiles + 7) / 8;
    const int pool_blocks  = (NN * 2 + 7) / 8;

    swizzle_weights<<<2 * LL, 256, 0, stream>>>(W1s, W2s, wswz);

    for (int l = 0; l < LL; l++) {
        const float* hsrc = (l == 0) ? x : h;
        copy_kernel<<<copy_blocks, 256, 0, stream>>>((const float4*)hsrc, (float4*)agg, n4);
        scatter_kernel<<<scatter_blks, 256, 0, stream>>>(hsrc, ei, agg);
        mlp_kernel<<<mlp_blocks, 256, 0, stream>>>(agg,
                                                   wswz + (size_t)(2 * l) * FF * FF,
                                                   wswz + (size_t)(2 * l + 1) * FF * FF,
                                                   h, ntiles);
    }
    zero_kernel<<<(BB * FF + 255) / 256, 256, 0, stream>>>(xr, BB * FF);
    pool_kernel<<<pool_blocks, 256, 0, stream>>>(h, batch, xr);
    head_kernel<<<1, 128, 0, stream>>>(xr, fcw, fcb, out);
}
